// MoBAAttention_52518860095896
// MI455X (gfx1250) — matrix operand order repaired and checked by reading, not compiled, not run on hardware
//
#include <hip/hip_runtime.h>

#define BB 2
#define SLEN 2048
#define HID 2048
#define NH 16
#define NKV 4
#define HD 128
#define CHK 256
#define NCHUNK 8
#define TOPK 4

typedef __attribute__((ext_vector_type(16))) __bf16 v16bf;
typedef __attribute__((ext_vector_type(8)))  __bf16 v8bf;
typedef __attribute__((ext_vector_type(8)))  float  v8f;
typedef __attribute__((ext_vector_type(4)))  unsigned int v4u;

static __device__ __forceinline__ unsigned short f2bf(float f) {
  unsigned u = __float_as_uint(f);
  u += 0x7FFFu + ((u >> 16) & 1u);
  return (unsigned short)(u >> 16);
}
static __device__ __forceinline__ float bf2f(unsigned short h) {
  return __uint_as_float(((unsigned)h) << 16);
}

static __device__ __forceinline__ v16bf cat8(v8bf lo, v8bf hi) {
  return __builtin_shufflevector(lo, hi, 0,1,2,3,4,5,6,7,8,9,10,11,12,13,14,15);
}

// A-matrix fragment (16xK slice, row-major source, ld in elements).
// ISA layout: lanes 0-15 row=lane, elems 0-7 = K+0..7, elems 8-15 = K+16..23;
// lanes 16-31 same row, K+8..15 / K+24..31.
static __device__ __forceinline__ v16bf load_fragA(const unsigned short* base, int row0, int ld, int k0) {
  int lane = (int)(threadIdx.x & 31u);
  const unsigned short* p = base + (size_t)(row0 + (lane & 15)) * ld + k0 + ((lane >> 4) << 3);
  v8bf lo = __builtin_bit_cast(v8bf, *(const v4u*)p);
  v8bf hi = __builtin_bit_cast(v8bf, *(const v4u*)(p + 16));
  return cat8(lo, hi);
}

// B-matrix fragment (Kx16), source stored transposed: col-major (N x K) with ld in elements.
// ISA layout: lanes 0-15 col=lane, elems 0-7 = K+0..7, elems 8-15 = K+16..23; lanes 16-31 K+8..15 / K+24..31.
static __device__ __forceinline__ v16bf load_fragB(const unsigned short* base, int col0, int ld, int k0) {
  int lane = (int)(threadIdx.x & 31u);
  const unsigned short* p = base + (size_t)(col0 + (lane & 15)) * ld + k0 + ((lane >> 4) << 3);
  v8bf lo = __builtin_bit_cast(v8bf, *(const v4u*)p);
  v8bf hi = __builtin_bit_cast(v8bf, *(const v4u*)(p + 16));
  return cat8(lo, hi);
}

static __device__ __forceinline__ v8f wmma_bf16(v16bf a, v16bf b, v8f c) {
  return __builtin_amdgcn_wmma_f32_16x16x32_bf16(false, a, false, b, (short)0, c, false, false);
}

// ---------------- elementwise prep kernels ----------------

__global__ void cvt_kernel(const float* __restrict__ s, unsigned short* __restrict__ d, int n) {
  int i = blockIdx.x * 256 + threadIdx.x;
  if (i < n) d[i] = f2bf(s[i]);
}

// d[n*K + k] = bf16(s[k*N + n])   (transpose K x N -> N x K)
__global__ void trans_kernel(const float* __restrict__ s, unsigned short* __restrict__ d, int K, int N) {
  long long tid = (long long)blockIdx.x * 256 + threadIdx.x;
  if (tid >= (long long)K * N) return;
  int k = (int)(tid % K);
  int nn = (int)(tid / K);
  d[tid] = f2bf(s[(size_t)k * N + nn]);
}

// src: [(b*S+s)][nheads*128] f32 ; dst: [b][h][s][128] bf16 with rope applied
__global__ void rope_kernel(const float* __restrict__ s, unsigned short* __restrict__ d, int nheads, int total) {
  int tid = blockIdx.x * 256 + threadIdx.x;
  if (tid >= total) return;
  int j = tid & 63;
  int r = tid >> 6;
  int hh = r % nheads; r /= nheads;
  int ss = r % SLEN;
  int b = r / SLEN;
  float invf = __powf(10000.0f, -(float)j * (1.0f / 64.0f));
  float ang = (float)ss * invf;
  float sn, cs;
  __sincosf(ang, &sn, &cs);
  size_t src = ((size_t)(b * SLEN + ss)) * (size_t)(nheads * HD) + (size_t)hh * HD + j;
  float x1 = s[src], x2 = s[src + 64];
  size_t dst = (((size_t)(b * nheads + hh)) * SLEN + ss) * HD + j;
  d[dst] = f2bf(x1 * cs - x2 * sn);
  d[dst + 64] = f2bf(x2 * cs + x1 * sn);
}

// v_f32: [(b*S+t)][512] ; vT: [b][kvh][d][t] bf16
__global__ void vtrans_kernel(const float* __restrict__ s, unsigned short* __restrict__ d, int total) {
  int tid = blockIdx.x * 256 + threadIdx.x;
  if (tid >= total) return;
  int t = tid & (SLEN - 1);
  int r = tid >> 11;
  int dd = r & (HD - 1); r >>= 7;
  int kvh = r & (NKV - 1);
  int b = r >> 2;
  d[tid] = f2bf(s[((size_t)(b * SLEN + t)) * (NKV * HD) + (size_t)kvh * HD + dd]);
}

// kmean[b][n][kvh][d] = mean over chunk of roped k
__global__ void kmean_kernel(const unsigned short* __restrict__ kb, float* __restrict__ km, int total) {
  int tid = blockIdx.x * 256 + threadIdx.x;
  if (tid >= total) return;
  int dd = tid & (HD - 1);
  int r = tid >> 7;
  int kvh = r & (NKV - 1);
  int nn = (r >> 2) & (NCHUNK - 1);
  int b = r >> 5;
  size_t base = (((size_t)(b * NKV + kvh)) * SLEN + (size_t)nn * CHK) * HD + dd;
  float acc = 0.f;
  for (int t = 0; t < CHK; ++t) acc += bf2f(kb[base + (size_t)t * HD]);
  km[tid] = acc * (1.0f / CHK);
}

// top-4 chunk selection per (b,h,s) -> bitmask
__global__ void gate_kernel(const unsigned short* __restrict__ qb, const float* __restrict__ km,
                            unsigned* __restrict__ amask, int total) {
  int tid = blockIdx.x * 256 + threadIdx.x;
  if (tid >= total) return;
  int ss = tid & (SLEN - 1);
  int h = (tid >> 11) & (NH - 1);
  int b = tid >> 15;
  int kvh = h >> 2;
  const unsigned short* qp = qb + (((size_t)(b * NH + h)) * SLEN + ss) * HD;
  float g[NCHUNK];
#pragma unroll
  for (int c = 0; c < NCHUNK; ++c) g[c] = 0.f;
  for (int dd = 0; dd < HD; ++dd) {
    float qd = bf2f(qp[dd]);
#pragma unroll
    for (int c = 0; c < NCHUNK; ++c)
      g[c] += qd * km[(((size_t)(b * NCHUNK + c)) * NKV + kvh) * HD + dd];
  }
  int qc = ss >> 8;
#pragma unroll
  for (int c = 0; c < NCHUNK; ++c) {
    if (c > qc) g[c] = -__builtin_inff();
    else if (c == qc) g[c] = __builtin_inff();
  }
  unsigned sel = 0;
#pragma unroll
  for (int p = 0; p < TOPK; ++p) {
    int bi = -1; float bv = 0.f;
#pragma unroll
    for (int c = 0; c < NCHUNK; ++c)
      if (!((sel >> c) & 1u) && (bi < 0 || g[c] > bv)) { bv = g[c]; bi = c; }
    sel |= 1u << bi;
  }
  amask[tid] = sel;
}

// ---------------- WMMA GEMM: C(MxN f32) = A(MxK bf16 rowmajor) x Bt(NxK bf16) ----------------

__global__ __launch_bounds__(256) void gemm_bf16_kernel(
    const unsigned short* __restrict__ A, const unsigned short* __restrict__ Bt,
    float* __restrict__ C, int M, int N, int K) {
  int wave = (int)(threadIdx.x >> 5);
  int lane = (int)(threadIdx.x & 31u);
  int half = lane >> 4, l15 = lane & 15;
  int wm = blockIdx.y * 128 + (wave >> 1) * 32;
  int wn = blockIdx.x * 128 + (wave & 1) * 64;
  v8f acc[2][4] = {};
  for (int k0 = 0; k0 < K; k0 += 32) {
    // high-locality prefetch (WGP scope: pulls into all cache levels)
    if (k0 + 32 < K) __builtin_prefetch(A + (size_t)wm * K + k0 + 32, 0, 3);
    v16bf a0 = load_fragA(A, wm, K, k0);
    v16bf a1 = load_fragA(A, wm + 16, K, k0);
    v16bf b0 = load_fragB(Bt, wn, K, k0);
    v16bf b1 = load_fragB(Bt, wn + 16, K, k0);
    v16bf b2 = load_fragB(Bt, wn + 32, K, k0);
    v16bf b3 = load_fragB(Bt, wn + 48, K, k0);
    acc[0][0] = wmma_bf16(a0, b0, acc[0][0]);
    acc[0][1] = wmma_bf16(a0, b1, acc[0][1]);
    acc[0][2] = wmma_bf16(a0, b2, acc[0][2]);
    acc[0][3] = wmma_bf16(a0, b3, acc[0][3]);
    acc[1][0] = wmma_bf16(a1, b0, acc[1][0]);
    acc[1][1] = wmma_bf16(a1, b1, acc[1][1]);
    acc[1][2] = wmma_bf16(a1, b2, acc[1][2]);
    acc[1][3] = wmma_bf16(a1, b3, acc[1][3]);
  }
#pragma unroll
  for (int mi = 0; mi < 2; ++mi)
#pragma unroll
    for (int ni = 0; ni < 4; ++ni)
#pragma unroll
      for (int v = 0; v < 8; ++v)
        C[(size_t)(wm + mi * 16 + v + half * 8) * N + (wn + ni * 16 + l15)] = acc[mi][ni][v];
}

// ---------------- block-sparse flash attention ----------------
// qb: [b][h][s][128] bf16; kb: [b][kvh][t][128] bf16; vtb: [b][kvh][d][t] bf16
// amask: [b][h][s] chunk bitmask; attn out: [(b*S+s)][h*128+d] bf16
__global__ __launch_bounds__(256) void attn_kernel(
    const unsigned short* __restrict__ qb, const unsigned short* __restrict__ kb,
    const unsigned short* __restrict__ vtb, const unsigned* __restrict__ amask,
    unsigned short* __restrict__ attn) {
  __shared__ unsigned short plds[8][16 * 32];
  int wg = blockIdx.x;
  int qblk = wg & 15, h = (wg >> 4) & 15, b = wg >> 8;
  int wave = (int)(threadIdx.x >> 5), lane = (int)(threadIdx.x & 31u);
  int half = lane >> 4, l15 = lane & 15;
  int s0 = qblk * 128 + wave * 16;
  int kvh = h >> 2;
  const unsigned short* qbase = qb + ((size_t)(b * NH + h)) * SLEN * HD;
  const unsigned short* kbase = kb + ((size_t)(b * NKV + kvh)) * SLEN * HD;
  const unsigned short* vbase = vtb + ((size_t)(b * NKV + kvh)) * HD * SLEN;

  v16bf qf[4];
#pragma unroll
  for (int kd = 0; kd < 4; ++kd) qf[kd] = load_fragA(qbase, s0, HD, kd * 32);

  unsigned m8[8];
  unsigned um = 0;
#pragma unroll
  for (int v = 0; v < 8; ++v) {
    m8[v] = amask[(size_t)(b * NH + h) * SLEN + s0 + v + half * 8];
    um |= m8[v];
  }
#pragma unroll
  for (int off = 1; off < 32; off <<= 1) um |= (unsigned)__shfl_xor((int)um, off, 32);

  float mrow[8], lrow[8], corr[8];
  v8f oacc[8] = {};
#pragma unroll
  for (int v = 0; v < 8; ++v) { mrow[v] = -3.0e38f; lrow[v] = 0.f; }

  int qchunk = s0 >> 8;
  const float scale = 0.088388347648318447f;  // 1/sqrt(128)

  for (int c = 0; c <= qchunk; ++c) {
    if (!((um >> c) & 1u)) continue;
    int cb = c << 8;
    int ngroups = (c == qchunk) ? (((s0 - cb) >> 5) + 1) : 8;
    for (int g = 0; g < ngroups; ++g) {
      int k0 = cb + g * 32;
      v8f sA = {}, sB = {};
#pragma unroll
      for (int kd = 0; kd < 4; ++kd) {
        v16bf kfA = load_fragB(kbase, k0, HD, kd * 32);
        v16bf kfB = load_fragB(kbase, k0 + 16, HD, kd * 32);
        sA = wmma_bf16(qf[kd], kfA, sA);
        sB = wmma_bf16(qf[kd], kfB, sB);
      }
#pragma unroll
      for (int v = 0; v < 8; ++v) {
        int srow = s0 + v + half * 8;
        int rowok = (int)((m8[v] >> c) & 1u);
        float a = sA[v] * scale;
        float bb2 = sB[v] * scale;
        if (!rowok || (k0 + l15) > srow) a = -3.0e38f;
        if (!rowok || (k0 + 16 + l15) > srow) bb2 = -3.0e38f;
        float gm = fmaxf(a, bb2);
#pragma unroll
        for (int off = 1; off < 16; off <<= 1) gm = fmaxf(gm, __shfl_xor(gm, off, 32));
        float mn = fmaxf(mrow[v], gm);
        float co, pa, pb;
        if (mn <= -1.0e37f) { co = 1.0f; pa = 0.0f; pb = 0.0f; }
        else { co = __expf(mrow[v] - mn); pa = __expf(a - mn); pb = __expf(bb2 - mn); mrow[v] = mn; }
        float rs = pa + pb;
#pragma unroll
        for (int off = 1; off < 16; off <<= 1) rs += __shfl_xor(rs, off, 32);
        lrow[v] = lrow[v] * co + rs;
        corr[v] = co;
        sA[v] = pa;
        sB[v] = pb;
      }
#pragma unroll
      for (int j = 0; j < 8; ++j)
#pragma unroll
        for (int v = 0; v < 8; ++v) oacc[j][v] *= corr[v];

      unsigned short* pl = &plds[wave][0];
#pragma unroll
      for (int v = 0; v < 8; ++v) {
        int row = v + half * 8;
        pl[row * 32 + l15] = f2bf(sA[v]);
        pl[row * 32 + 16 + l15] = f2bf(sB[v]);
      }
      asm volatile("s_wait_dscnt 0" ::: "memory");
      v16bf pf = load_fragA(pl, 0, 32, 0);
#pragma unroll
      for (int j = 0; j < 8; ++j) {
        v16bf vf = load_fragB(vbase, j * 16, SLEN, k0);
        oacc[j] = wmma_bf16(pf, vf, oacc[j]);
      }
    }
  }
#pragma unroll
  for (int v = 0; v < 8; ++v) lrow[v] = (lrow[v] > 0.f) ? (1.0f / lrow[v]) : 0.f;
#pragma unroll
  for (int j = 0; j < 8; ++j)
#pragma unroll
    for (int v = 0; v < 8; ++v) {
      int s = s0 + v + half * 8;
      attn[(size_t)(b * SLEN + s) * (NH * HD) + h * HD + j * 16 + l15] = f2bf(oacc[j][v] * lrow[v]);
    }
}

// ---------------- host launcher ----------------

extern "C" void kernel_launch(void* const* d_in, const int* in_sizes, int n_in,
                              void* d_out, int out_size, void* d_ws, size_t ws_size,
                              hipStream_t stream) {
  (void)in_sizes; (void)n_in; (void)out_size; (void)ws_size;
  const float* hs = (const float*)d_in[0];
  const float* wq = (const float*)d_in[1];
  const float* wk = (const float*)d_in[2];
  const float* wv = (const float*)d_in[3];
  const float* wo = (const float*)d_in[4];
  float* out = (float*)d_out;
  char* ws = (char*)d_ws;
  size_t off = 0;
  auto alloc = [&](size_t bytes) -> void* {
    void* p = ws + off;
    off += (bytes + 255) & ~(size_t)255;
    return p;
  };
  const int BS = BB * SLEN;  // 4096
  unsigned short* hid_bf = (unsigned short*)alloc((size_t)BS * HID * 2);  // reused as attn-out later
  unsigned short* wqT = (unsigned short*)alloc((size_t)2048 * 2048 * 2);
  unsigned short* wkT = (unsigned short*)alloc((size_t)512 * 2048 * 2);
  unsigned short* wvT = (unsigned short*)alloc((size_t)512 * 2048 * 2);
  unsigned short* woT = (unsigned short*)alloc((size_t)2048 * 2048 * 2);
  float* qf32 = (float*)alloc((size_t)BS * 2048 * 4);
  float* kf32 = (float*)alloc((size_t)BS * 512 * 4);
  float* vf32 = (float*)alloc((size_t)BS * 512 * 4);
  unsigned short* qbf = (unsigned short*)alloc((size_t)BB * NH * SLEN * HD * 2);
  unsigned short* kbf = (unsigned short*)alloc((size_t)BB * NKV * SLEN * HD * 2);
  unsigned short* vT = (unsigned short*)alloc((size_t)BB * NKV * HD * SLEN * 2);
  float* km = (float*)alloc((size_t)BB * NCHUNK * NKV * HD * 4);
  unsigned* amask = (unsigned*)alloc((size_t)BB * NH * SLEN * 4);

  int n;
  n = BS * HID;
  cvt_kernel<<<n / 256, 256, 0, stream>>>(hs, hid_bf, n);
  n = 2048 * 2048;
  trans_kernel<<<n / 256, 256, 0, stream>>>(wq, wqT, 2048, 2048);
  n = 2048 * 512;
  trans_kernel<<<n / 256, 256, 0, stream>>>(wk, wkT, 2048, 512);
  trans_kernel<<<n / 256, 256, 0, stream>>>(wv, wvT, 2048, 512);
  n = 2048 * 2048;
  trans_kernel<<<n / 256, 256, 0, stream>>>(wo, woT, 2048, 2048);

  gemm_bf16_kernel<<<dim3(2048 / 128, BS / 128), 256, 0, stream>>>(hid_bf, wqT, qf32, BS, 2048, 2048);
  gemm_bf16_kernel<<<dim3(512 / 128, BS / 128), 256, 0, stream>>>(hid_bf, wkT, kf32, BS, 512, 2048);
  gemm_bf16_kernel<<<dim3(512 / 128, BS / 128), 256, 0, stream>>>(hid_bf, wvT, vf32, BS, 512, 2048);

  n = BB * SLEN * NH * 64;
  rope_kernel<<<n / 256, 256, 0, stream>>>(qf32, qbf, NH, n);
  n = BB * SLEN * NKV * 64;
  rope_kernel<<<n / 256, 256, 0, stream>>>(kf32, kbf, NKV, n);
  n = BB * NKV * HD * SLEN;
  vtrans_kernel<<<n / 256, 256, 0, stream>>>(vf32, vT, n);
  n = BB * NCHUNK * NKV * HD;
  kmean_kernel<<<(n + 255) / 256, 256, 0, stream>>>(kbf, km, n);
  n = BB * NH * SLEN;
  gate_kernel<<<n / 256, 256, 0, stream>>>(qbf, km, amask, n);

  attn_kernel<<<BB * NH * (SLEN / 128), 256, 0, stream>>>(qbf, kbf, vT, amask, hid_bf);

  gemm_bf16_kernel<<<dim3(2048 / 128, BS / 128), 256, 0, stream>>>(hid_bf, woT, out, BS, 2048, 2048);
}
